// ROIHead_80951543595268
// MI455X (gfx1250) — matrix operand order repaired and checked by reading, not compiled, not run on hardware
//
#include <hip/hip_runtime.h>
#include <hip/hip_bf16.h>
#include <math.h>

typedef __attribute__((ext_vector_type(16))) __bf16 v16bf;
typedef __attribute__((ext_vector_type(8)))  float  v8f;
typedef unsigned long long u64;
typedef unsigned int  u32;
typedef unsigned short u16;

#define BATCH 8
#define IMH   288
#define IMW   512
#define TH    8
#define TW    32
#define BLKX  (IMW / TW)       // 16
#define BLKY  (IMH / TH)       // 36
#define NBLK  (BLKX * BLKY)    // 576 blocks per image
#define NCAND (NBLK * 2)       // 1152 candidates per image
#define NFRAG1 72              // 9 taps * 2 ksteps * 4 ntiles
#define NFRAG2 36              // 9 taps * 2 ksteps * 2 ntiles
#define CPAD  72               // channel stride (144B): bank-conflict-free

union FragAB {
  v16bf v;
  uint4 q[2];
};

__device__ __forceinline__ u16 f2bf(float f) {
  u32 u = __float_as_uint(f);
  u = (u + 0x7FFFu + ((u >> 16) & 1u)) >> 16;  // round-to-nearest-even
  return (u16)u;
}

// ---------------------------------------------------------------------------
// Kernel 0: pack OIHW fp32 weights into per-lane bf16 WMMA B-fragments.
// 16-bit B 32x16 (ISA 7.12.2): lane L holds column N=L%16; lanes 0-15 hold
// K in {0..7} U {16..23}; lanes 16-31 hold K in {8..15} U {24..31}.
// Stored per-lane-contiguous (32B) for single-clause loads.
// ---------------------------------------------------------------------------
__global__ void pack_weights(const float* __restrict__ w1,
                             const float* __restrict__ w2,
                             u16* __restrict__ pw1, u16* __restrict__ pw2) {
  int e = blockIdx.x * blockDim.x + threadIdx.x;
  const int N1 = NFRAG1 * 512;
  const int N2 = NFRAG2 * 512;
  if (e < N1) {
    int f = e >> 9, r = e & 511;
    int lane = r >> 4, el = r & 15;
    int n = f & 3, g = f >> 2;
    int s = g & 1, tap = g >> 1;
    int dy = tap / 3, dx = tap % 3;
    int half = (lane < 16) ? 0 : 8;
    int K = (el < 8) ? (el + half) : (8 + el + half);
    int ci = s * 32 + K;
    int co = n * 16 + (lane & 15);
    pw1[e] = f2bf(w1[((co * 64 + ci) * 3 + dy) * 3 + dx]);
  } else if (e < N1 + N2) {
    int e2 = e - N1;
    int f = e2 >> 9, r = e2 & 511;
    int lane = r >> 4, el = r & 15;
    int n = f & 1, g = f >> 1;
    int s = g & 1, tap = g >> 1;
    int dy = tap / 3, dx = tap % 3;
    int half = (lane < 16) ? 0 : 8;
    int K = (el < 8) ? (el + half) : (8 + el + half);
    int ci = s * 32 + K;
    int co = n * 16 + (lane & 15);
    pw2[e2] = f2bf(w2[((co * 64 + ci) * 3 + dy) * 3 + dx]);
  }
}

// ---------------------------------------------------------------------------
// Kernel 1: fully fused conv1(relu) -> conv2(relu) -> conv3(sigmoid) -> top-2.
// One block = one 8x32 heatmap tile of one image. 256 threads = 8 wave32.
// Each wave owns TWO output-channel tiles -> each A fragment load feeds two
// WMMAs (1 ds_load_b128 per v_wmma). Channel stride padded to 72 (144B) so
// the 16-lane A gathers hit all 64 LDS banks exactly twice (no conflicts).
// ---------------------------------------------------------------------------
__global__ __launch_bounds__(256)
void fused_head(const float* __restrict__ x,
                const u16* __restrict__ pw1, const float* __restrict__ b1,
                const u16* __restrict__ pw2, const float* __restrict__ b2,
                const float* __restrict__ w3, const float* __restrict__ b3,
                u64* __restrict__ cand) {
  // input tile: rows r0-2..r0+9, cols c0-9..c0+40 (bf16, channel-contiguous)
  __shared__ __align__(16) u16 sIn[12][50][CPAD];  // 86.4 KB
  // conv1 output: rows r0-1..r0+8, cols c0-8..c0+39
  __shared__ __align__(16) u16 sC1[10][48][CPAD];  // 69.1 KB
  __shared__ float sHeat[TH * TW];                 // 1 KB
  __shared__ u64 sKeys[256];                       // 2 KB

  const int tid  = threadIdx.x;
  const int lane = tid & 31;
  const int wv   = tid >> 5;
  const int m    = lane & 15;
  const int half = (lane >> 4) << 3;  // 0 for lanes 0-15, 8 for lanes 16-31

  const int cx = blockIdx.x, cy = blockIdx.y, b = blockIdx.z;
  const int c0 = cx * TW, r0 = cy * TH;

  // ---- stage 0: global fp32 -> LDS bf16 input tile (zero-padded) ----
  const int TOT = 12 * 50 * 64;
  for (int idx = tid; idx < TOT; idx += 256) {
    int col = idx % 50;
    int t   = idx / 50;
    int ch  = t & 63;
    int rr  = t >> 6;
    int gr = r0 - 2 + rr;
    int gc = c0 - 9 + col;
    float v = 0.0f;
    if ((u32)gr < (u32)IMH && (u32)gc < (u32)IMW)
      v = x[(((size_t)b * 64 + ch) * IMH + gr) * IMW + gc];
    sIn[rr][col][ch] = f2bf(v);
  }

  // ---- conv1 B-fragments: this wave's ntile PAIR, pinned in VGPRs ----
  const int p1 = wv & 1;              // pair select: ntiles {2p1, 2p1+1}
  FragAB wB0[18], wB1[18];
#pragma unroll
  for (int fi = 0; fi < 18; ++fi) {
    const u16* pa = pw1 + ((size_t)((fi << 2) + 2 * p1)     * 32 + lane) * 16;
    const u16* pb = pw1 + ((size_t)((fi << 2) + 2 * p1 + 1) * 32 + lane) * 16;
    wB0[fi].q[0] = *(const uint4*)pa;
    wB0[fi].q[1] = *(const uint4*)(pa + 8);
    wB1[fi].q[0] = *(const uint4*)pb;
    wB1[fi].q[1] = *(const uint4*)(pb + 8);
  }
  const int ch0 = 2 * p1 * 16 + m;
  const float bias1a = b1[ch0];
  const float bias1b = b1[ch0 + 16];
  __syncthreads();

  // ---- conv1: 10 rows x 3 M-tiles; each A load feeds 2 WMMAs ----
  for (int q = (wv >> 1); q < 30; q += 4) {
    int rr = q / 3, t = q % 3;
    v8f acc0 = {}, acc1 = {};
#pragma unroll
    for (int dy = 0; dy < 3; ++dy)
#pragma unroll
      for (int dx = 0; dx < 3; ++dx)
#pragma unroll
        for (int s = 0; s < 2; ++s) {
          const u16* ap = &sIn[rr + dy][16 * t + m + dx][s * 32 + half];
          FragAB A;
          A.q[0] = *(const uint4*)ap;
          A.q[1] = *(const uint4*)(ap + 16);
          const int i = (dy * 3 + dx) * 2 + s;
          acc0 = __builtin_amdgcn_wmma_f32_16x16x32_bf16(
              false, A.v, false, wB0[i].v, (short)0, acc0, false, false);
          acc1 = __builtin_amdgcn_wmma_f32_16x16x32_bf16(
              false, A.v, false, wB1[i].v, (short)0, acc1, false, false);
        }
#pragma unroll
    for (int v = 0; v < 8; ++v) {
      float y0 = acc0[v] + bias1a;
      float y1 = acc1[v] + bias1b;
      y0 = y0 > 0.0f ? y0 : 0.0f;
      y1 = y1 > 0.0f ? y1 : 0.0f;
      sC1[rr][16 * t + v + half][ch0]      = f2bf(y0);  // D: M = v + 8*(lane/16)
      sC1[rr][16 * t + v + half][ch0 + 16] = f2bf(y1);
    }
  }

  // ---- conv2 B-fragments: BOTH ntiles (all 32 channels) per wave ----
  FragAB wC0[18], wC1[18];
#pragma unroll
  for (int fi = 0; fi < 18; ++fi) {
    const u16* pa = pw2 + ((size_t)((fi << 1) + 0) * 32 + lane) * 16;
    const u16* pb = pw2 + ((size_t)((fi << 1) + 1) * 32 + lane) * 16;
    wC0[fi].q[0] = *(const uint4*)pa;
    wC0[fi].q[1] = *(const uint4*)(pa + 8);
    wC1[fi].q[0] = *(const uint4*)pb;
    wC1[fi].q[1] = *(const uint4*)(pb + 8);
  }
  const float bias2a = b2[m];
  const float bias2b = b2[16 + m];
  const float w3a    = w3[m];
  const float w3b    = w3[16 + m];
  __syncthreads();

  // ---- conv2 (+relu) fused with conv3: full 32-ch reduction in-wave ----
  for (int q = wv; q < 16; q += 8) {
    int rr = q >> 1, t2 = q & 1;
    v8f acc0 = {}, acc1 = {};
#pragma unroll
    for (int dy = 0; dy < 3; ++dy)
#pragma unroll
      for (int dx = 0; dx < 3; ++dx)
#pragma unroll
        for (int s = 0; s < 2; ++s) {
          const u16* ap = &sC1[rr + dy][16 * t2 + m + dx + 7][s * 32 + half];
          FragAB A;
          A.q[0] = *(const uint4*)ap;
          A.q[1] = *(const uint4*)(ap + 16);
          const int i = (dy * 3 + dx) * 2 + s;
          acc0 = __builtin_amdgcn_wmma_f32_16x16x32_bf16(
              false, A.v, false, wC0[i].v, (short)0, acc0, false, false);
          acc1 = __builtin_amdgcn_wmma_f32_16x16x32_bf16(
              false, A.v, false, wC1[i].v, (short)0, acc1, false, false);
        }
#pragma unroll
    for (int v = 0; v < 8; ++v) {
      float y0 = acc0[v] + bias2a;
      float y1 = acc1[v] + bias2b;
      y0 = y0 > 0.0f ? y0 : 0.0f;
      y1 = y1 > 0.0f ? y1 : 0.0f;
      float part = y0 * w3a + y1 * w3b;        // conv3 1x1 contribution
      part += __shfl_xor(part, 1, 32);         // reduce 16 lanes = 32 channels
      part += __shfl_xor(part, 2, 32);
      part += __shfl_xor(part, 4, 32);
      part += __shfl_xor(part, 8, 32);
      if (m == 0)                              // lanes 0 and 16: unique owner
        sHeat[rr * TW + 16 * t2 + v + half] = part;
    }
  }
  __syncthreads();

  // ---- sigmoid + pack sortable key (value bits << 32 | ~index) ----
  {
    int row = tid >> 5, col = tid & 31;
    int gr = r0 + row, gc = c0 + col;
    float hv = sHeat[tid] + b3[0];
    float sg = 1.0f / (1.0f + __expf(-hv));    // in (0,1): bits are monotonic
    u32 gidx = (u32)(gr * IMW + gc);
    sKeys[tid] = ((u64)__float_as_uint(sg) << 32) | (u64)(0xFFFFFFFFu - gidx);
  }
  __syncthreads();

  if (tid == 0) {
    u64 best = 0, second = 0;
#pragma unroll 1
    for (int i = 0; i < 256; ++i) {
      u64 k = sKeys[i];
      if (k > best)        { second = best; best = k; }
      else if (k > second) { second = k; }
    }
    int blk = blockIdx.y * gridDim.x + blockIdx.x;
    u64* o = cand + ((size_t)b * NBLK + blk) * 2;
    o[0] = best;
    o[1] = second;
  }
}

// ---------------------------------------------------------------------------
// Kernel 2: reduce per-block candidates to global top-2 per image, emit ROIs.
// Output: rois (9, 16, 4) = [b, t, h, w].
// ---------------------------------------------------------------------------
__global__ void finalize(const u64* __restrict__ cand, float* __restrict__ out) {
  __shared__ u64 sK[512];
  const int b = blockIdx.x, tid = threadIdx.x;
  u64 best = 0, second = 0;
  for (int i = tid; i < NCAND; i += 256) {
    u64 k = cand[(size_t)b * NCAND + i];
    if (k > best)        { second = best; best = k; }
    else if (k > second) { second = k; }
  }
  sK[tid * 2]     = best;
  sK[tid * 2 + 1] = second;
  __syncthreads();
  if (tid == 0) {
    u64 B0 = 0, B1 = 0;
    for (int i = 0; i < 512; ++i) {
      u64 k = sK[i];
      if (k > B0)      { B1 = B0; B0 = k; }
      else if (k > B1) { B1 = k; }
    }
    sK[0] = B0;
    sK[1] = B1;
  }
  __syncthreads();
  if (tid < 9) {
    int t = tid;
    for (int i = 0; i < 2; ++i) {
      u32 gidx = 0xFFFFFFFFu - (u32)sK[i];
      float h = (float)(gidx / IMW);
      float w = (float)(gidx % IMW);
      int j = b * 2 + i;
      float* o = out + ((size_t)t * 16 + j) * 4;
      o[0] = (float)b;
      o[1] = (float)t;
      o[2] = h;
      o[3] = w;
    }
  }
}

// ---------------------------------------------------------------------------
extern "C" void kernel_launch(void* const* d_in, const int* in_sizes, int n_in,
                              void* d_out, int out_size, void* d_ws, size_t ws_size,
                              hipStream_t stream) {
  const float* x  = (const float*)d_in[0];
  const float* w1 = (const float*)d_in[1];
  const float* b1 = (const float*)d_in[2];
  const float* w2 = (const float*)d_in[3];
  const float* b2 = (const float*)d_in[4];
  const float* w3 = (const float*)d_in[5];
  const float* b3 = (const float*)d_in[6];
  float* out = (float*)d_out;

  // workspace layout: [candidates 73728B][pw1 73728B][pw2 36864B]
  u64* cand = (u64*)d_ws;
  u16* pw1  = (u16*)((char*)d_ws + (size_t)BATCH * NCAND * sizeof(u64));
  u16* pw2  = pw1 + NFRAG1 * 512;

  const int packN = (NFRAG1 + NFRAG2) * 512;
  pack_weights<<<(packN + 255) / 256, 256, 0, stream>>>(w1, w2, pw1, pw2);

  dim3 grid(BLKX, BLKY, BATCH);
  fused_head<<<grid, 256, 0, stream>>>(x, pw1, b1, pw2, b2, w3, b3, cand);

  finalize<<<BATCH, 256, 0, stream>>>(cand, out);
}
